// Encoder_18854906430072
// MI455X (gfx1250) — compile-verified, matrix operand order checked by reading, not run on hardware
//
#include <hip/hip_runtime.h>
#include <hip/hip_bf16.h>
#include <math.h>

typedef __attribute__((ext_vector_type(16))) _Float16 v16h;
typedef __attribute__((ext_vector_type(8)))  float    v8f;
typedef __attribute__((ext_vector_type(4)))  float    f4;

#define NUM_CLASSES 4
#define VMAX 64
#define HID 64
#define OUT 768
#define IN_VS 28
#define KEYS (NUM_CLASSES * VMAX)   // 256
#define CNT_SZ (KEYS * IN_VS)       // 7168
#define WAVES 8

// ---------------- zero scratch counters ----------------
__global__ void k_zero(unsigned* p, int n) {
    int i = blockIdx.x * blockDim.x + threadIdx.x;
    if (i < n) p[i] = 0u;
}

// ---------------- W_fc -> f16 ----------------
__global__ void k_wfc_h(const float* __restrict__ wfc, _Float16* __restrict__ out, int n) {
    int i = blockIdx.x * blockDim.x + threadIdx.x;
    if (i < n) out[i] = (_Float16)wfc[i];
}

// ---------------- (class,value,col) histogram, LDS-privatized ----------------
__global__ __launch_bounds__(256) void k_hist(const float* __restrict__ VS,
                                              const int* __restrict__ Level,
                                              unsigned* __restrict__ counts,
                                              unsigned* __restrict__ csize, int N) {
    __shared__ unsigned lc[CNT_SZ];
    __shared__ unsigned lcs[NUM_CLASSES];
    for (int i = threadIdx.x; i < CNT_SZ; i += blockDim.x) lc[i] = 0u;
    if (threadIdx.x < NUM_CLASSES) lcs[threadIdx.x] = 0u;
    __syncthreads();
    for (int row = blockIdx.x * blockDim.x + threadIdx.x; row < N;
         row += gridDim.x * blockDim.x) {
        int lab = Level[row];
        atomicAdd(&lcs[lab], 1u);
        const float* vr = VS + (size_t)row * IN_VS;
        #pragma unroll
        for (int c = 0; c < IN_VS; ++c) {
            int v = (int)vr[c];
            atomicAdd(&lc[(lab * VMAX + v) * IN_VS + c], 1u);
        }
    }
    __syncthreads();
    for (int i = threadIdx.x; i < CNT_SZ; i += blockDim.x) {
        unsigned x = lc[i];
        if (x) atomicAdd(&counts[i], x);
    }
    if (threadIdx.x < NUM_CLASSES) {
        unsigned x = lcs[threadIdx.x];
        if (x) atomicAdd(&csize[threadIdx.x], x);
    }
}

// ---------------- CC copy: streaming, non-temporal both ways ----------------
__global__ void k_copy(const f4* __restrict__ src, f4* __restrict__ dst, long n4) {
    long i = (long)blockIdx.x * blockDim.x + threadIdx.x;
    long stride = (long)gridDim.x * blockDim.x;
    for (; i < n4; i += stride) {
        if (i + stride < n4) __builtin_prefetch(&src[i + stride], 0, 0);
        f4 v = __builtin_nontemporal_load(&src[i]);
        __builtin_nontemporal_store(v, &dst[i]);
    }
}

// ---------------- Level/Depart -> float tail of d_out ----------------
__global__ void k_ld(const int* __restrict__ L, const int* __restrict__ D,
                     float* __restrict__ oL, float* __restrict__ oD, int N) {
    int i = blockIdx.x * blockDim.x + threadIdx.x;
    if (i < N) { oL[i] = (float)L[i]; oD[i] = (float)D[i]; }
}

// ---------------- fused: casual_weighted + LSTM gates + WMMA fc ----------------
__global__ __launch_bounds__(256) void k_main(const float* __restrict__ VS,
                                              const int* __restrict__ Level,
                                              const float* __restrict__ Wih,
                                              const float* __restrict__ bih,
                                              const float* __restrict__ bhh,
                                              const _Float16* __restrict__ WfcH,
                                              const float* __restrict__ bfc,
                                              const unsigned* __restrict__ counts,
                                              const unsigned* __restrict__ csize,
                                              float* __restrict__ out_vs, int N) {
    __shared__ float    sW[256 * IN_VS];        // W_ih staged, [256][28]
    __shared__ float    sB[256];                // b_ih + b_hh
    __shared__ float    sBF[OUT];               // b_fc staged
    __shared__ float    sX[WAVES][16 * IN_VS];  // vs_cw tiles [16][28]
    __shared__ _Float16 sH[WAVES][16 * 72];     // h tiles, stride 72 halves

    const int tid  = threadIdx.x;
    const int w    = tid >> 5;      // wave32
    const int lane = tid & 31;

    // stage W_ih + bias sums + b_fc (block-wide)
    for (int i = tid; i < 256 * IN_VS; i += 256) sW[i] = Wih[i];
    sB[tid] = bih[tid] + bhh[tid];
    for (int i = tid; i < OUT; i += 256) sBF[i] = bfc[i];
    __syncthreads();

    const int tile = blockIdx.x * WAVES + w;
    const int r0   = tile * 16;

    // ---- phase 1: vs_cw tile (448 elements / 32 lanes) ----
    #pragma unroll
    for (int t = 0; t < 14; ++t) {
        int e = lane + 32 * t;           // 0..447
        int r = e / IN_VS;
        int c = e - r * IN_VS;
        int row = r0 + r; if (row >= N) row = N - 1;   // clamp for tail tiles
        int lab = Level[row];
        float v = VS[(size_t)row * IN_VS + c];
        unsigned cnt = counts[(lab * VMAX + (int)v) * IN_VS + c];
        float cs = (float)csize[lab];
        sX[w][r * IN_VS + c] = v * ((float)cnt / cs);
    }
    __syncthreads();

    // ---- phase 2: gates (i,g,o only; f is dead since c0=0) -> h (f16) ----
    for (int r = 0; r < 16; ++r) {
        float x[IN_VS];
        #pragma unroll
        for (int k = 0; k < IN_VS; ++k) x[k] = sX[w][r * IN_VS + k];  // broadcast row
        #pragma unroll
        for (int jj = 0; jj < 2; ++jj) {
            int j = lane + 32 * jj;      // hidden unit 0..63
            float gi = sB[j], gg = sB[128 + j], go = sB[192 + j];
            const float* wi = &sW[j * IN_VS];
            const float* wg = &sW[(128 + j) * IN_VS];
            const float* wo = &sW[(192 + j) * IN_VS];
            #pragma unroll
            for (int k = 0; k < IN_VS; ++k) {
                gi = fmaf(x[k], wi[k], gi);
                gg = fmaf(x[k], wg[k], gg);
                go = fmaf(x[k], wo[k], go);
            }
            float c_ = (1.f / (1.f + __expf(-gi))) * tanhf(gg);
            float h_ = (1.f / (1.f + __expf(-go))) * tanhf(c_);
            sH[w][r * 72 + j] = (_Float16)h_;
        }
    }
    __syncthreads();

    // ---- phase 3: vs_feat = h @ W_fc^T + b_fc via v_wmma_f32_16x16x32_f16 ----
    // A (16x32 f16) lane layout: m = lane&15; k-half = lane>>4;
    //   elements 0..7 -> K = 8*kh + 0..7 ; elements 8..15 -> K = 16 + 8*kh + 0..7
    const int m     = lane & 15;
    const int kbase = (lane >> 4) * 8;
    const _Float16* hrow = &sH[w][m * 72];
    v16h a0, a1;
    #pragma unroll
    for (int e = 0; e < 8; ++e) {
        a0[e]     = hrow[kbase + e];            // kk=0, K =  kbase+e
        a0[8 + e] = hrow[16 + kbase + e];       //        K = 16+kbase+e
        a1[e]     = hrow[32 + kbase + e];       // kk=1
        a1[8 + e] = hrow[48 + kbase + e];
    }

    const int mrow = (lane >> 4) * 8;           // D: lanes>=16 hold M=8..15
    const bool full = (r0 + 16 <= N);           // wave-uniform tail test

    if (full) {
        // fast path: no store guards, non-temporal streaming stores
        float* obase = out_vs + (size_t)(r0 + mrow) * OUT + m;
        for (int nt = 0; nt < 48; ++nt) {
            int n = nt * 16 + m;
            const _Float16* brow = WfcH + (size_t)n * HID;
            v16h b0, b1;
            #pragma unroll
            for (int e = 0; e < 8; ++e) {
                b0[e]     = brow[kbase + e];
                b0[8 + e] = brow[16 + kbase + e];
                b1[e]     = brow[32 + kbase + e];
                b1[8 + e] = brow[48 + kbase + e];
            }
            v8f acc = {};
            acc = __builtin_amdgcn_wmma_f32_16x16x32_f16(false, a0, false, b0,
                                                         (short)0, acc, false, false);
            acc = __builtin_amdgcn_wmma_f32_16x16x32_f16(false, a1, false, b1,
                                                         (short)0, acc, false, false);
            float bias = sBF[n];
            float* o = obase + nt * 16;
            #pragma unroll
            for (int v = 0; v < 8; ++v)
                __builtin_nontemporal_store(acc[v] + bias, o + (size_t)v * OUT);
        }
    } else {
        // tail tile: per-row guards
        for (int nt = 0; nt < 48; ++nt) {
            int n = nt * 16 + m;
            const _Float16* brow = WfcH + (size_t)n * HID;
            v16h b0, b1;
            #pragma unroll
            for (int e = 0; e < 8; ++e) {
                b0[e]     = brow[kbase + e];
                b0[8 + e] = brow[16 + kbase + e];
                b1[e]     = brow[32 + kbase + e];
                b1[8 + e] = brow[48 + kbase + e];
            }
            v8f acc = {};
            acc = __builtin_amdgcn_wmma_f32_16x16x32_f16(false, a0, false, b0,
                                                         (short)0, acc, false, false);
            acc = __builtin_amdgcn_wmma_f32_16x16x32_f16(false, a1, false, b1,
                                                         (short)0, acc, false, false);
            float bias = sBF[n];
            #pragma unroll
            for (int v = 0; v < 8; ++v) {
                int row = r0 + mrow + v;
                if (row < N) out_vs[(size_t)row * OUT + n] = acc[v] + bias;
            }
        }
    }
}

extern "C" void kernel_launch(void* const* d_in, const int* in_sizes, int n_in,
                              void* d_out, int out_size, void* d_ws, size_t ws_size,
                              hipStream_t stream) {
    const float* VS   = (const float*)d_in[0];
    const float* CC   = (const float*)d_in[1];
    const int*   Lev  = (const int*)d_in[2];
    const int*   Dep  = (const int*)d_in[3];
    const float* Wih  = (const float*)d_in[4];
    // d_in[5] = W_hh: mathematically dead (h0 = 0)
    const float* bih  = (const float*)d_in[6];
    const float* bhh  = (const float*)d_in[7];
    const float* Wfc  = (const float*)d_in[8];
    const float* bfc  = (const float*)d_in[9];
    const int N = in_sizes[0] / IN_VS;

    // scratch layout
    unsigned* counts = (unsigned*)d_ws;                       // 7168 u32
    unsigned* csize  = counts + CNT_SZ;                       // 4 u32
    _Float16* WfcH   = (_Float16*)((char*)d_ws + 32768);      // 768*64 f16

    // output layout: cc_feat [N*768] | vs_feat [N*768] | Level [N] | Depart [N]
    float* out_cc = (float*)d_out;
    float* out_vs = out_cc + (size_t)N * OUT;
    float* out_L  = out_vs + (size_t)N * OUT;
    float* out_D  = out_L + N;

    k_zero<<<(CNT_SZ + NUM_CLASSES + 255) / 256, 256, 0, stream>>>(counts, CNT_SZ + NUM_CLASSES);
    k_wfc_h<<<(OUT * HID + 255) / 256, 256, 0, stream>>>(Wfc, WfcH, OUT * HID);
    k_hist<<<256, 256, 0, stream>>>(VS, Lev, counts, csize, N);

    long n4 = (long)N * OUT / 4;
    k_copy<<<8192, 256, 0, stream>>>((const f4*)CC, (f4*)out_cc, n4);
    k_ld<<<(N + 255) / 256, 256, 0, stream>>>(Lev, Dep, out_L, out_D, N);

    int tiles  = (N + 15) / 16;
    int blocks = (tiles + WAVES - 1) / WAVES;
    k_main<<<blocks, 256, 0, stream>>>(VS, Lev, Wih, bih, bhh, WfcH, bfc,
                                       counts, csize, out_vs, N);
}
